// CMPNN_51823075393627
// MI455X (gfx1250) — compile-verified, matrix operand order checked by reading, not run on hardware
//
#include <hip/hip_runtime.h>
#include <hip/hip_bf16.h>
#include <math.h>

// ---------------------------------------------------------------- constants
#define NN   16384   // nodes
#define BB   256     // graphs
#define NPG  64      // nodes per graph
#define DEG  8       // in-degree (dst = repeat(arange(N), DEG) -> edges of node n are [8n, 8n+8))
#define EE   (NN*DEG)
#define HH   256
#define FN   133
#define FNP  160     // FN padded to mult of 32
#define FE   14
#define FEP  32      // FE padded to mult of 32

typedef __bf16 bf16;
typedef __attribute__((ext_vector_type(16))) __bf16 v16bf;
typedef __attribute__((ext_vector_type(8)))  float  v8f;

// ---------------------------------------------------------------- WMMA GEMM
// C[M,N] = act(A[M,K] @ Bt[N,K]^T + bias). A/Bt bf16 row-major, K % 32 == 0,
// M % 128 == 0, N % 64 == 0 (true for every GEMM in this network).
//
// Blocking: one 256-thread block (8 waves) owns a 128x64 C tile. The 64xK_c
// B tile is staged in LDS once per block (8-way inter-wave reuse; padded row
// stride of 136 elems -> 272B -> bank advance 4/row -> conflict-free
// ds_load_b128 fragment reads). Each wave computes a 16x64 strip with 4
// v8f accumulators, so each A fragment feeds 4 back-to-back
// v_wmma_f32_16x16x32_bf16 ops.
//
// Fragment layout per CDNA5 ISA (wave32), 16-bit A 16x32:
//   lanes 0-15: row = lane, K {0..7} in v0..3, K {16..23} in v4..7
//   lanes 16-31: same row,  K {8..15},         K {24..31}
// -> two contiguous 16B loads per operand per K-chunk.
//
// Compile-time WRITE_BF / RELU keep the epilogue branch-free (bias always
// present; exactly one of f32 / bf16 output is written per call).
#define KC    128     // K staging chunk
#define BPAD  136     // LDS row stride (128 + 8 pad)

template <bool WRITE_BF, bool RELU>
__global__ void gemm_bf16_wmma(const bf16* __restrict__ A, const bf16* __restrict__ Bt,
                               const float* __restrict__ bias,
                               float* __restrict__ Cf, bf16* __restrict__ Cb,
                               int M, int N, int K)
{
    __shared__ bf16 sB[64 * BPAD];

    int tid   = threadIdx.x;
    int w     = tid >> 5;          // wave 0..7
    int lane  = tid & 31;
    int lidx  = lane & 15;
    int lhalf = lane >> 4;
    int ntN = N >> 6;
    int tm  = blockIdx.x / ntN;    // 128-row block
    int tn  = blockIdx.x % ntN;    // 64-col block

    const bf16* arow = A + (size_t)(tm * 128 + w * 16 + lidx) * K + lhalf * 8;

    v8f acc[4];
#pragma unroll
    for (int j = 0; j < 4; ++j) acc[j] = (v8f){0.f, 0.f, 0.f, 0.f, 0.f, 0.f, 0.f, 0.f};

    for (int k0 = 0; k0 < K; k0 += KC) {
        int kc = K - k0; if (kc > KC) kc = KC;
        int nch = kc >> 3;                       // 16B chunks per B row

        __syncthreads();                         // previous iteration's reads done
        for (int c = tid; c < 64 * nch; c += 256) {
            int row = c / nch;
            int ko  = (c - row * nch) * 8;
            *(uint4*)&sB[row * BPAD + ko] =
                *(const uint4*)&Bt[(size_t)(tn * 64 + row) * K + k0 + ko];
        }
        __syncthreads();

        __builtin_prefetch(arow + k0 + KC, 0, 3);   // global_prefetch_b8: next A chunk

        for (int kk = 0; kk < kc; kk += 32) {
            union { v16bf v; uint4 u[2]; } fa;
            fa.u[0] = *(const uint4*)(arow + k0 + kk);
            fa.u[1] = *(const uint4*)(arow + k0 + kk + 16);
#pragma unroll
            for (int j = 0; j < 4; ++j) {
                union { v16bf v; uint4 u[2]; } fb;
                const bf16* bp = &sB[(j * 16 + lidx) * BPAD + kk + lhalf * 8];
                fb.u[0] = *(const uint4*)(bp);
                fb.u[1] = *(const uint4*)(bp + 16);
                acc[j] = __builtin_amdgcn_wmma_f32_16x16x32_bf16(
                             false, fa.v, false, fb.v, (short)0, acc[j], false, false);
            }
        }
    }

    int rbase = tm * 128 + w * 16 + lhalf * 8;   // C/D layout: vgpr r, lanes16-31 -> M+8
#pragma unroll
    for (int j = 0; j < 4; ++j) {
        int col = tn * 64 + j * 16 + lidx;
        float bv = bias[col];
#pragma unroll
        for (int r = 0; r < 8; ++r) {
            float v = acc[j][r] + bv;
            if (RELU) v = fmaxf(v, 0.f);
            size_t off = (size_t)(rbase + r) * N + col;
            if (WRITE_BF) Cb[off] = (bf16)v;
            else          Cf[off] = v;
        }
    }
}

// ---------------------------------------------------------------- helpers
__global__ void k_cvt_pad(const float* __restrict__ in, bf16* __restrict__ out,
                          int M, int K, int Kp)
{
    size_t i = (size_t)blockIdx.x * blockDim.x + threadIdx.x;
    if (i >= (size_t)M * Kp) return;
    int k = (int)(i % Kp);
    size_t m = i / Kp;
    out[i] = (k < K) ? (bf16)in[m * K + k] : (bf16)0.f;
}

// W[K,N] row-major -> Wt[N,Kp] row-major bf16 (zero-padded K)
__global__ void k_transpose_w(const float* __restrict__ W, bf16* __restrict__ Wt,
                              int K, int N, int Kp)
{
    size_t i = (size_t)blockIdx.x * blockDim.x + threadIdx.x;
    if (i >= (size_t)N * Kp) return;
    int k = (int)(i % Kp);
    int n = (int)(i / Kp);
    Wt[i] = (k < K) ? (bf16)W[(size_t)k * N + n] : (bf16)0.f;
}

// Fused edge-recompute + segment sum/max + node update. Block = one node,
// thread = one feature. h_e element = t ? relu(ie + t[src]) : ie.
// mode A (mout==null): hn_out = hn_in + s*mx (also bf16 copy)
// mode B (mout!=null): mout = s*mx
__global__ void k_mpnn_reduce(const bf16* __restrict__ ie, const float* t,
                              const int* __restrict__ src,
                              const float* hn_in, float* hn_out, bf16* hn_bf,
                              float* mout)
{
    __shared__ int ssrc[DEG];
    int n = blockIdx.x;
    int h = threadIdx.x;
    if (t && h < DEG) ssrc[h] = src[n * DEG + h];
    __syncthreads();

    float s = 0.f, mx = -3.0e38f;
    for (int j = 0; j < DEG; ++j) {
        size_t e = (size_t)n * DEG + j;
        float v = (float)ie[e * HH + h];
        if (t) v = fmaxf(v + t[(size_t)ssrc[j] * HH + h], 0.f);
        s += v;
        mx = fmaxf(mx, v);
    }
    float m = s * mx;
    size_t idx = (size_t)n * HH + h;
    if (mout) {
        mout[idx] = m;
    } else {
        float hv = hn_in[idx] + m;
        hn_out[idx] = hv;
        hn_bf[idx]  = (bf16)hv;
    }
}

// cat = [m_final, h_n, i_n] as bf16 [N, 3H]
__global__ void k_pack_cat(const float* __restrict__ mfin, const float* __restrict__ hn,
                           const float* __restrict__ inode, bf16* __restrict__ cat)
{
    size_t i = (size_t)blockIdx.x * blockDim.x + threadIdx.x;
    if (i >= (size_t)NN * 3 * HH) return;
    int c = (int)(i % (3 * HH));
    size_t n = i / (3 * HH);
    float v;
    if      (c < HH)     v = mfin [n * HH + c];
    else if (c < 2 * HH) v = hn   [n * HH + c - HH];
    else                 v = inode[n * HH + c - 2 * HH];
    cat[i] = (bf16)v;
}

__global__ void k_msg(const float* __restrict__ hlin, const float* __restrict__ gbias,
                      bf16* __restrict__ msg)
{
    size_t i = (size_t)blockIdx.x * blockDim.x + threadIdx.x;
    if (i >= (size_t)NN * HH) return;
    msg[i] = (bf16)fmaxf(hlin[i] + gbias[i % HH], 0.f);
}

// h0 = per-graph max of hlin (pre-activation), init both GRU hiddens + zero accumulators
__global__ void k_h0(const float* __restrict__ hlin,
                     float* hf_f, bf16* hf_b, float* hb_f, bf16* hb_b,
                     float* gaccf, float* gaccb)
{
    int i = blockIdx.x * blockDim.x + threadIdx.x;
    if (i >= BB * HH) return;
    int b = i / HH, h = i % HH;
    float mx = -3.0e38f;
    for (int j = 0; j < NPG; ++j)
        mx = fmaxf(mx, hlin[((size_t)b * NPG + j) * HH + h]);
    hf_f[i] = mx; hf_b[i] = (bf16)mx;
    hb_f[i] = mx; hb_b[i] = (bf16)mx;
    gaccf[i] = 0.f; gaccb[i] = 0.f;
}

__device__ __forceinline__ float sigm(float x) { return 1.f / (1.f + expf(-x)); }

// One GRU step for one direction. xg = msg@Wi+bi precomputed [NN,3H], gh = h@Wh+bh [BB,3H].
// node_t selects the sequence position (fwd: t, bwd: 63-t). Mean-pool accumulated on the fly
// (mean over positions == mean over steps, so the bwd double-reversal is irrelevant).
__global__ void k_gru_step(const float* __restrict__ xg, const float* __restrict__ gh,
                           float* h_f, bf16* h_b, float* gacc, int node_t)
{
    int i = blockIdx.x * blockDim.x + threadIdx.x;
    if (i >= BB * HH) return;
    int b = i / HH, h = i % HH;
    const float* x = xg + ((size_t)b * NPG + node_t) * (3 * HH);
    const float* g = gh + (size_t)b * (3 * HH);
    float r  = sigm(x[h]          + g[h]);
    float z  = sigm(x[HH + h]     + g[HH + h]);
    float nn = tanhf(x[2 * HH + h] + r * g[2 * HH + h]);
    float hn = (1.f - z) * nn + z * h_f[i];
    h_f[i] = hn;
    h_b[i] = (bf16)hn;
    gacc[i] += hn;
}

__global__ void k_gcat(const float* __restrict__ gaccf, const float* __restrict__ gaccb,
                       bf16* __restrict__ gcat)
{
    int i = blockIdx.x * blockDim.x + threadIdx.x;
    if (i >= BB * 2 * HH) return;
    int b = i / (2 * HH), c = i % (2 * HH);
    float v = (c < HH ? gaccf[b * HH + c] : gaccb[b * HH + c - HH]) * (1.f / NPG);
    gcat[i] = (bf16)v;
}

__global__ void k_out(const bf16* __restrict__ ffn, const float* __restrict__ W2,
                      const float* __restrict__ b2, float* __restrict__ out)
{
    int b = blockIdx.x * blockDim.x + threadIdx.x;
    if (b >= BB) return;
    float s = 0.f;
    for (int h = 0; h < HH; ++h) s += (float)ffn[(size_t)b * HH + h] * W2[h];
    out[b] = s + b2[0];
}

// ---------------------------------------------------------------- launchers
static inline int cdiv(long long a, long long b) { return (int)((a + b - 1) / b); }

// f32 output
static void gemm_f(hipStream_t s, const bf16* A, const bf16* Bt, const float* bias,
                   float* Cf, int M, int N, int K, int relu)
{
    int blocks = (M / 128) * (N / 64);       // exact tiling: M%128==0, N%64==0
    if (relu)
        gemm_bf16_wmma<false, true ><<<blocks, 256, 0, s>>>(A, Bt, bias, Cf, nullptr, M, N, K);
    else
        gemm_bf16_wmma<false, false><<<blocks, 256, 0, s>>>(A, Bt, bias, Cf, nullptr, M, N, K);
}

// bf16 output (always relu in this network)
static void gemm_b(hipStream_t s, const bf16* A, const bf16* Bt, const float* bias,
                   bf16* Cb, int M, int N, int K)
{
    int blocks = (M / 128) * (N / 64);
    gemm_bf16_wmma<true, true><<<blocks, 256, 0, s>>>(A, Bt, bias, nullptr, Cb, M, N, K);
}

extern "C" void kernel_launch(void* const* d_in, const int* in_sizes, int n_in,
                              void* d_out, int out_size, void* d_ws, size_t ws_size,
                              hipStream_t stream)
{
    const float* node_x = (const float*)d_in[0];
    const float* edge_x = (const float*)d_in[1];
    const int*   src    = (const int*)  d_in[2];
    /* d_in[3] = dst is structurally repeat(arange(N), DEG) -> edges of node n are [8n,8n+8) */
    const float* Wn  = (const float*)d_in[4];  const float* bn  = (const float*)d_in[5];
    const float* We  = (const float*)d_in[6];  const float* be  = (const float*)d_in[7];
    const float* Wm0 = (const float*)d_in[8];  const float* bm0 = (const float*)d_in[9];
    const float* Wm1 = (const float*)d_in[10]; const float* bm1 = (const float*)d_in[11];
    const float* Wlr = (const float*)d_in[12]; const float* blr = (const float*)d_in[13];
    const float* gbias = (const float*)d_in[14];
    const float* Wif = (const float*)d_in[15]; const float* Whf = (const float*)d_in[16];
    const float* bif = (const float*)d_in[17]; const float* bhf = (const float*)d_in[18];
    const float* Wib = (const float*)d_in[19]; const float* Whb = (const float*)d_in[20];
    const float* bib = (const float*)d_in[21]; const float* bhb = (const float*)d_in[22];
    const float* W1  = (const float*)d_in[23]; const float* b1  = (const float*)d_in[24];
    const float* W2  = (const float*)d_in[25]; const float* b2  = (const float*)d_in[26];
    float* out = (float*)d_out;

    // bump allocator over d_ws (deterministic layout every call)
    char* ws = (char*)d_ws;
    size_t cur = 0;
    auto alloc = [&](size_t bytes) -> void* {
        cur = (cur + 255) & ~(size_t)255;
        void* p = ws + cur;
        cur += bytes;
        return p;
    };

    bf16* nodeA = (bf16*)alloc((size_t)NN * FNP * 2);
    bf16* edgeA = (bf16*)alloc((size_t)EE * FEP * 2);
    bf16* WnT   = (bf16*)alloc((size_t)HH * FNP * 2);
    bf16* WeT   = (bf16*)alloc((size_t)HH * FEP * 2);
    bf16* Wm0T  = (bf16*)alloc((size_t)HH * HH * 2);
    bf16* Wm1T  = (bf16*)alloc((size_t)HH * HH * 2);
    bf16* WlrT  = (bf16*)alloc((size_t)HH * 3 * HH * 2);
    bf16* WifT  = (bf16*)alloc((size_t)3 * HH * HH * 2);
    bf16* WhfT  = (bf16*)alloc((size_t)3 * HH * HH * 2);
    bf16* WibT  = (bf16*)alloc((size_t)3 * HH * HH * 2);
    bf16* WhbT  = (bf16*)alloc((size_t)3 * HH * HH * 2);
    bf16* W1T   = (bf16*)alloc((size_t)HH * 2 * HH * 2);
    float* in_f  = (float*)alloc((size_t)NN * HH * 4);
    bf16*  ie_bf = (bf16*) alloc((size_t)EE * HH * 2);
    float* hn_f  = (float*)alloc((size_t)NN * HH * 4);
    bf16*  hn_bf = (bf16*) alloc((size_t)NN * HH * 2);
    float* t_f   = (float*)alloc((size_t)NN * HH * 4);
    float* mfin  = (float*)alloc((size_t)NN * HH * 4);
    bf16*  cat_bf= (bf16*) alloc((size_t)NN * 3 * HH * 2);
    float* hlin  = (float*)alloc((size_t)NN * HH * 4);
    bf16*  msg_bf= (bf16*) alloc((size_t)NN * HH * 2);
    float* xgf   = (float*)alloc((size_t)NN * 3 * HH * 4);
    float* xgb   = (float*)alloc((size_t)NN * 3 * HH * 4);
    float* hf_f  = (float*)alloc((size_t)BB * HH * 4);
    bf16*  hf_b  = (bf16*) alloc((size_t)BB * HH * 2);
    float* hb_f  = (float*)alloc((size_t)BB * HH * 4);
    bf16*  hb_b  = (bf16*) alloc((size_t)BB * HH * 2);
    float* ghf   = (float*)alloc((size_t)BB * 3 * HH * 4);
    float* ghb   = (float*)alloc((size_t)BB * 3 * HH * 4);
    float* gaccf = (float*)alloc((size_t)BB * HH * 4);
    float* gaccb = (float*)alloc((size_t)BB * HH * 4);
    bf16*  gcat  = (bf16*) alloc((size_t)BB * 2 * HH * 2);
    bf16*  ffn_bf= (bf16*) alloc((size_t)BB * HH * 2);
    (void)ws_size; (void)in_sizes; (void)n_in; (void)out_size;

    const int TB = 256;
    // ---- bf16 conversions / weight transposes (B stored [N,Kp] for WMMA B-operand)
    k_cvt_pad<<<cdiv((size_t)NN * FNP, TB), TB, 0, stream>>>(node_x, nodeA, NN, FN, FNP);
    k_cvt_pad<<<cdiv((size_t)EE * FEP, TB), TB, 0, stream>>>(edge_x, edgeA, EE, FE, FEP);
    k_transpose_w<<<cdiv((size_t)HH * FNP, TB), TB, 0, stream>>>(Wn,  WnT,  FN,  HH,   FNP);
    k_transpose_w<<<cdiv((size_t)HH * FEP, TB), TB, 0, stream>>>(We,  WeT,  FE,  HH,   FEP);
    k_transpose_w<<<cdiv((size_t)HH * HH,  TB), TB, 0, stream>>>(Wm0, Wm0T, HH,  HH,   HH);
    k_transpose_w<<<cdiv((size_t)HH * HH,  TB), TB, 0, stream>>>(Wm1, Wm1T, HH,  HH,   HH);
    k_transpose_w<<<cdiv((size_t)HH * 3*HH,TB), TB, 0, stream>>>(Wlr, WlrT, 3*HH, HH,  3*HH);
    k_transpose_w<<<cdiv((size_t)3*HH * HH,TB), TB, 0, stream>>>(Wif, WifT, HH,  3*HH, HH);
    k_transpose_w<<<cdiv((size_t)3*HH * HH,TB), TB, 0, stream>>>(Whf, WhfT, HH,  3*HH, HH);
    k_transpose_w<<<cdiv((size_t)3*HH * HH,TB), TB, 0, stream>>>(Wib, WibT, HH,  3*HH, HH);
    k_transpose_w<<<cdiv((size_t)3*HH * HH,TB), TB, 0, stream>>>(Whb, WhbT, HH,  3*HH, HH);
    k_transpose_w<<<cdiv((size_t)HH * 2*HH,TB), TB, 0, stream>>>(W1,  W1T,  2*HH, HH,  2*HH);

    // ---- embeddings
    gemm_f(stream, nodeA, WnT, bn, in_f, NN, HH, FNP, 1);   // i_n = relu(node_x@Wn+bn)
    gemm_b(stream, edgeA, WeT, be, ie_bf, EE, HH, FEP);     // i_e = relu(edge_x@We+be)

    // ---- message passing (h_e recomputed on the fly; (h_n[src])@W == (h_n@W)[src])
    k_mpnn_reduce<<<NN, HH, 0, stream>>>(ie_bf, nullptr, src, in_f, hn_f, hn_bf, nullptr);
    gemm_f(stream, hn_bf, Wm0T, bm0, t_f, NN, HH, HH, 0);
    k_mpnn_reduce<<<NN, HH, 0, stream>>>(ie_bf, t_f, src, hn_f, hn_f, hn_bf, nullptr);
    gemm_f(stream, hn_bf, Wm1T, bm1, t_f, NN, HH, HH, 0);
    k_mpnn_reduce<<<NN, HH, 0, stream>>>(ie_bf, t_f, src, nullptr, nullptr, nullptr, mfin);

    // ---- readout linear over concat
    k_pack_cat<<<cdiv((size_t)NN * 3 * HH, TB), TB, 0, stream>>>(mfin, hn_f, in_f, cat_bf);
    gemm_f(stream, cat_bf, WlrT, blr, hlin, NN, HH, 3 * HH, 0);

    // ---- BatchGRU
    k_msg<<<cdiv((size_t)NN * HH, TB), TB, 0, stream>>>(hlin, gbias, msg_bf);
    k_h0 <<<cdiv(BB * HH, TB), TB, 0, stream>>>(hlin, hf_f, hf_b, hb_f, hb_b, gaccf, gaccb);
    gemm_f(stream, msg_bf, WifT, bif, xgf, NN, 3 * HH, HH, 0);  // input gates, fwd
    gemm_f(stream, msg_bf, WibT, bib, xgb, NN, 3 * HH, HH, 0);  // input gates, bwd

    for (int t = 0; t < NPG; ++t) {
        gemm_f(stream, hf_b, WhfT, bhf, ghf, BB, 3 * HH, HH, 0);
        k_gru_step<<<cdiv(BB * HH, TB), TB, 0, stream>>>(xgf, ghf, hf_f, hf_b, gaccf, t);
        gemm_f(stream, hb_b, WhbT, bhb, ghb, BB, 3 * HH, HH, 0);
        k_gru_step<<<cdiv(BB * HH, TB), TB, 0, stream>>>(xgb, ghb, hb_f, hb_b, gaccb, NPG - 1 - t);
    }

    // ---- FFN head
    k_gcat<<<cdiv(BB * 2 * HH, TB), TB, 0, stream>>>(gaccf, gaccb, gcat);
    gemm_b(stream, gcat, W1T, b1, ffn_bf, BB, HH, 2 * HH);
    k_out<<<1, BB, 0, stream>>>(ffn_bf, W2, b2, out);
}
